// SLA_A_83502754169337
// MI455X (gfx1250) — matrix operand order repaired and checked by reading, not compiled, not run on hardware
//
#include <hip/hip_runtime.h>
#include <hip/hip_bf16.h>
#include <math.h>

typedef __attribute__((ext_vector_type(16))) _Float16 v16h;
typedef __attribute__((ext_vector_type(8)))  float    v8f;

// ---------------- WMMA fragment helpers (wave32, 16x16x32 f16) ----------------
// A fragment: lane L -> M = L%16 ; element s holds K = 16*(L/16) + s (contiguous, matches B)
__device__ __forceinline__ v16h frag_a_lds(const _Float16* buf, int stride, int m0, int k0, int lane) {
  int hg = (lane >> 4) & 1;
  int m  = m0 + (lane & 15);
  v16h a;
#pragma unroll
  for (int j = 0; j < 8; ++j) {
    int kb = 2 * j + 16 * hg;
    a[2 * j]     = buf[m * stride + k0 + kb];
    a[2 * j + 1] = buf[m * stride + k0 + kb + 1];
  }
  return a;
}
// B fragment from LDS stored as [K][N]
__device__ __forceinline__ v16h frag_b_kn_lds(const _Float16* buf, int stride, int k0, int n0, int lane) {
  int hg = (lane >> 4) & 1;
  int n  = n0 + (lane & 15);
  v16h b;
#pragma unroll
  for (int e = 0; e < 16; ++e) b[e] = buf[(k0 + 16 * hg + e) * stride + n];
  return b;
}
// B fragment from LDS stored transposed as [N][K]
__device__ __forceinline__ v16h frag_b_nk_lds(const _Float16* buf, int stride, int k0, int n0, int lane) {
  int hg = (lane >> 4) & 1;
  int n  = n0 + (lane & 15);
  const _Float16* p = buf + n * stride + k0 + 16 * hg;
  v16h b;
#pragma unroll
  for (int e = 0; e < 16; ++e) b[e] = p[e];
  return b;
}
// B fragment from GLOBAL fp32 weights stored [N][K] (e.g. torch Linear weight), convert to f16
__device__ __forceinline__ v16h frag_b_nk_gf32(const float* buf, int stride, int k0, int n0, int lane) {
  int hg = (lane >> 4) & 1;
  int n  = n0 + (lane & 15);
  const float* p = buf + n * stride + k0 + 16 * hg;
  v16h b;
#pragma unroll
  for (int e = 0; e < 16; ++e) b[e] = (_Float16)p[e];
  return b;
}
__device__ __forceinline__ v8f wmma_f16(v16h a, v16h b, v8f c) {
  return __builtin_amdgcn_wmma_f32_16x16x32_f16(false, a, false, b, (short)0, c, false, false);
}

// B=8, C=64, H=W=256, WIN=8, X=Y=32, windows n = 8192
#define ATTN_OFF 33554432  // 8*64*256*256

// =====================================================================
// Kernel 1: window partition + LN + single-head 8x8-window attention
// one block (4 waves, 128 thr) per window; writes x2 (ws) and attn (out)
// =====================================================================
__global__ __launch_bounds__(128) void swin_attn_kernel(
    const float* __restrict__ x, const float* __restrict__ ln_w, const float* __restrict__ ln_b,
    const float* __restrict__ qkv_w, const float* __restrict__ out_w,
    float* __restrict__ x2, float* __restrict__ attn_out)
{
  __shared__ float    xs[64 * 65];       // raw window (residual), stride 65 -> conflict-free LN
  __shared__ _Float16 bufA[64 * 72];     // phase 1-2: xn ; phase 3-4: attn
  __shared__ _Float16 bufB[64 * 72];     // phase 2-3: q (pre-scaled) ; phase 4-5: o
  __shared__ _Float16 kh[64 * 72];
  __shared__ _Float16 vh[64 * 72];

  const int tid  = threadIdx.x;
  const int lane = tid & 31;
  const int w    = tid >> 5;             // wave 0..3
  const int hg   = (lane >> 4) & 1;
  const int win  = blockIdx.x;           // n = b*1024 + wx*32 + wy
  const int b    = win >> 10;
  const int wx   = (win >> 5) & 31;
  const int wy   = win & 31;
  const int row0 = wx * 8, col0 = wy * 8;

  // ---- load window: xs[t][c] = x[b][c][row0+t/8][col0+t%8] (8-contig segments) ----
#pragma unroll
  for (int i = 0; i < 32; ++i) {
    int f = i * 128 + tid;               // 0..4095
    int c = f >> 6, t = f & 63;
    xs[t * 65 + c] = x[(((size_t)b * 64 + c) * 256 + row0 + (t >> 3)) * 256 + col0 + (t & 7)];
  }
  __syncthreads();

  // ---- per-token LayerNorm over channels (eps 1e-5) -> bufA (f16) ----
  if (tid < 64) {
    int t = tid;
    float s = 0.f, s2 = 0.f;
#pragma unroll
    for (int c = 0; c < 64; ++c) { float v = xs[t * 65 + c]; s += v; s2 += v * v; }
    float mu   = s * (1.f / 64.f);
    float var  = s2 * (1.f / 64.f) - mu * mu;
    float rstd = rsqrtf(var + 1e-5f);
#pragma unroll
    for (int c = 0; c < 64; ++c)
      bufA[t * 72 + c] = (_Float16)(((xs[t * 65 + c] - mu) * rstd) * ln_w[c] + ln_b[c]);
  }
  __syncthreads();

  // ---- QKV: wave w computes token rows 16w..16w+15 of [64 x 192] ----
  {
    v16h a0 = frag_a_lds(bufA, 72, 16 * w, 0, lane);
    v16h a1 = frag_a_lds(bufA, 72, 16 * w, 32, lane);
#pragma unroll
    for (int ni = 0; ni < 12; ++ni) {
      v16h b0 = frag_b_nk_gf32(qkv_w, 64, 0,  16 * ni, lane);
      v16h b1 = frag_b_nk_gf32(qkv_w, 64, 32, 16 * ni, lane);
      v8f acc = {};
      acc = wmma_f16(a0, b0, acc);
      acc = wmma_f16(a1, b1, acc);
      float mul = (ni < 4) ? 0.125f : 1.0f;           // fold scale = C^-0.5 into q
      _Float16* dst = (ni < 4) ? bufB : ((ni < 8) ? kh : vh);
      int n0 = 16 * (ni & 3);
#pragma unroll
      for (int r = 0; r < 8; ++r)
        dst[(16 * w + r + 8 * hg) * 72 + n0 + (lane & 15)] = (_Float16)(acc[r] * mul);
    }
  }
  __syncthreads();

  // ---- sim = q @ k^T ; rowwise softmax ; write attn (f32 global + f16 LDS) ----
  {
    v16h a0 = frag_a_lds(bufB, 72, 16 * w, 0, lane);
    v16h a1 = frag_a_lds(bufB, 72, 16 * w, 32, lane);
    v8f acc[4];
#pragma unroll
    for (int ni = 0; ni < 4; ++ni) {
      v8f c = {};
      c = wmma_f16(a0, frag_b_nk_lds(kh, 72, 0,  16 * ni, lane), c);  // B[k=d][n=j] = k[j][d]
      c = wmma_f16(a1, frag_b_nk_lds(kh, 72, 32, 16 * ni, lane), c);
      acc[ni] = c;
    }
    float* ag = attn_out + (size_t)win * 4096;
#pragma unroll
    for (int r = 0; r < 8; ++r) {
      float e0 = acc[0][r], e1 = acc[1][r], e2 = acc[2][r], e3 = acc[3][r];
      float mx = fmaxf(fmaxf(e0, e1), fmaxf(e2, e3));
      mx = fmaxf(mx, __shfl_xor(mx, 1, 32));
      mx = fmaxf(mx, __shfl_xor(mx, 2, 32));
      mx = fmaxf(mx, __shfl_xor(mx, 4, 32));
      mx = fmaxf(mx, __shfl_xor(mx, 8, 32));          // reduce within 16-lane N-group
      e0 = __expf(e0 - mx); e1 = __expf(e1 - mx); e2 = __expf(e2 - mx); e3 = __expf(e3 - mx);
      float s = e0 + e1 + e2 + e3;
      s += __shfl_xor(s, 1, 32);
      s += __shfl_xor(s, 2, 32);
      s += __shfl_xor(s, 4, 32);
      s += __shfl_xor(s, 8, 32);
      float inv = 1.0f / s;
      e0 *= inv; e1 *= inv; e2 *= inv; e3 *= inv;
      int m = 16 * w + r + 8 * hg;
      int n = lane & 15;
      ag[m * 64 + n]      = e0; ag[m * 64 + 16 + n] = e1;
      ag[m * 64 + 32 + n] = e2; ag[m * 64 + 48 + n] = e3;
      bufA[m * 72 + n]      = (_Float16)e0; bufA[m * 72 + 16 + n] = (_Float16)e1;
      bufA[m * 72 + 32 + n] = (_Float16)e2; bufA[m * 72 + 48 + n] = (_Float16)e3;
    }
  }
  __syncthreads();

  // ---- o = attn @ v -> bufB ----
  {
    v16h a0 = frag_a_lds(bufA, 72, 16 * w, 0, lane);
    v16h a1 = frag_a_lds(bufA, 72, 16 * w, 32, lane);
#pragma unroll
    for (int ni = 0; ni < 4; ++ni) {
      v8f c = {};
      c = wmma_f16(a0, frag_b_kn_lds(vh, 72, 0,  16 * ni, lane), c);  // B[k=j][n=d] = v[j][d]
      c = wmma_f16(a1, frag_b_kn_lds(vh, 72, 32, 16 * ni, lane), c);
#pragma unroll
      for (int r = 0; r < 8; ++r)
        bufB[(16 * w + r + 8 * hg) * 72 + 16 * ni + (lane & 15)] = (_Float16)c[r];
    }
  }
  __syncthreads();

  // ---- final = o @ out_w^T + residual (accumulate into xs) ----
  {
    v16h a0 = frag_a_lds(bufB, 72, 16 * w, 0, lane);
    v16h a1 = frag_a_lds(bufB, 72, 16 * w, 32, lane);
#pragma unroll
    for (int ni = 0; ni < 4; ++ni) {
      v8f c = {};
      c = wmma_f16(a0, frag_b_nk_gf32(out_w, 64, 0,  16 * ni, lane), c);
      c = wmma_f16(a1, frag_b_nk_gf32(out_w, 64, 32, 16 * ni, lane), c);
#pragma unroll
      for (int r = 0; r < 8; ++r) {
        int m = 16 * w + r + 8 * hg, n = 16 * ni + (lane & 15);
        xs[m * 65 + n] += c[r];
      }
    }
  }
  __syncthreads();

  // ---- coalesced write of x2 to workspace ----
#pragma unroll
  for (int i = 0; i < 32; ++i) {
    int f = i * 128 + tid;
    int c = f >> 6, t = f & 63;
    x2[(((size_t)b * 64 + c) * 256 + row0 + (t >> 3)) * 256 + col0 + (t & 7)] = xs[t * 65 + c];
  }
}

// =====================================================================
// Kernel 2: LayerNorm2d + 1x1(64->128) + dw3x3 + GELU-gate + 1x1(64) + residual
// one block (8 waves, 256 thr) per 8x8 pixel tile (10x10 halo)
// =====================================================================
__global__ __launch_bounds__(256) void swin_ffn_kernel(
    const float* __restrict__ x2, const float* __restrict__ cn_w, const float* __restrict__ cn_b,
    const float* __restrict__ pin_w, const float* __restrict__ dw_w, const float* __restrict__ pout_w,
    float* __restrict__ out)
{
  // LDS carve (aliased regions), total 60928 B
  __shared__ __align__(16) unsigned char smem[30464 + 16640 + 9216 + 4608];
  float*    x2t  = (float*)smem;                        // [100][69] f32 (27600B), phase A
  _Float16* ph   = (_Float16*)smem;                     // [112][136] f16 (30464B), phase B+ (aliases x2t)
  _Float16* xch  = (_Float16*)(smem + 30464);           // [112][72] f16 (16128B), phase A-B
  float*    outt = (float*)(smem + 30464);              // [64][65] f32 (16640B), phase D+ (aliases xch)
  _Float16* gh   = (_Float16*)(smem + 30464 + 16640);   // [64][72] f16 (9216B)
  float*    dww  = (float*)(smem + 30464 + 16640 + 9216); // [128*9] f32

  const int tid  = threadIdx.x;
  const int lane = tid & 31;
  const int w    = tid >> 5;            // wave 0..7
  const int hg   = (lane >> 4) & 1;
  const int blk  = blockIdx.x;          // b*1024 + ty*32 + tx
  const int b    = blk >> 10;
  const int ty   = (blk >> 5) & 31;
  const int tx   = blk & 31;
  const int r0   = ty * 8, c0 = tx * 8;

  // stage depthwise weights
  for (int f = tid; f < 1152; f += 256) dww[f] = dw_w[f];

  // load 10x10 halo of x2 (zero outside image)
  for (int f = tid; f < 6400; f += 256) {
    int c = f / 100, pix = f % 100;
    int hr = r0 - 1 + pix / 10, hc = c0 - 1 + (pix % 10);
    float v = 0.f;
    if (hr >= 0 && hr < 256 && hc >= 0 && hc < 256)
      v = x2[(((size_t)b * 64 + c) * 256 + hr) * 256 + hc];
    x2t[pix * 69 + c] = v;
  }
  // zero M-pad rows 100..111 of xc
  for (int f = tid; f < 12 * 72; f += 256) xch[100 * 72 + f] = (_Float16)0.f;
  __syncthreads();

  // LayerNorm2d per pixel (eps 1e-6); invalid halo -> xc = 0 so p = 0 (zero-pad conv)
  if (tid < 100) {
    int pix = tid;
    int hr = r0 - 1 + pix / 10, hc = c0 - 1 + (pix % 10);
    bool valid = (hr >= 0 && hr < 256 && hc >= 0 && hc < 256);
    float s = 0.f, s2 = 0.f;
#pragma unroll
    for (int c = 0; c < 64; ++c) { float v = x2t[pix * 69 + c]; s += v; s2 += v * v; }
    float mu   = s * (1.f / 64.f);
    float var  = s2 * (1.f / 64.f) - mu * mu;
    float rstd = rsqrtf(var + 1e-6f);
#pragma unroll
    for (int c = 0; c < 64; ++c) {
      float v = valid ? (cn_w[c] * ((x2t[pix * 69 + c] - mu) * rstd) + cn_b[c]) : 0.f;
      xch[pix * 72 + c] = (_Float16)v;
    }
  }
  __syncthreads();

  // pin GEMM: p[112 x 128] = xc @ pin_w^T ; wave w owns output column tile w
  {
    v16h b0 = frag_b_nk_gf32(pin_w, 64, 0,  16 * w, lane);
    v16h b1 = frag_b_nk_gf32(pin_w, 64, 32, 16 * w, lane);
#pragma unroll
    for (int mi = 0; mi < 7; ++mi) {
      v16h a0 = frag_a_lds(xch, 72, 16 * mi, 0, lane);
      v16h a1 = frag_a_lds(xch, 72, 16 * mi, 32, lane);
      v8f c = {};
      c = wmma_f16(a0, b0, c);
      c = wmma_f16(a1, b1, c);
#pragma unroll
      for (int r = 0; r < 8; ++r) {
        int m = 16 * mi + r + 8 * hg;
        ph[m * 136 + 16 * w + (lane & 15)] = (_Float16)c[r];
      }
    }
  }
  __syncthreads();

  // depthwise 3x3 + exact GELU gate (fp32), interior 8x8 pixels
  for (int it = 0; it < 16; ++it) {
    int idx = it * 256 + tid;           // 64 pixels * 64 ch
    int pix = idx >> 6, c = idx & 63;
    int li = pix >> 3, lj = pix & 7;
    float dlo = 0.f, dhi = 0.f;
#pragma unroll
    for (int di = 0; di < 3; ++di)
#pragma unroll
      for (int dj = 0; dj < 3; ++dj) {
        int hp = (li + di) * 10 + (lj + dj);
        dlo += (float)ph[hp * 136 + c]      * dww[c * 9 + di * 3 + dj];
        dhi += (float)ph[hp * 136 + c + 64] * dww[(c + 64) * 9 + di * 3 + dj];
      }
    float g = 0.5f * dlo * (1.0f + erff(dlo * 0.70710678118654752f)) * dhi;
    gh[pix * 72 + c] = (_Float16)g;
  }
  __syncthreads();

  // pout GEMM: out[64 x 64] = g @ pout_w^T ; 16 tiles over 8 waves
#pragma unroll
  for (int t = 0; t < 2; ++t) {
    int tile = w * 2 + t;
    int mi = tile >> 2, ni = tile & 3;
    v16h a0 = frag_a_lds(gh, 72, 16 * mi, 0, lane);
    v16h a1 = frag_a_lds(gh, 72, 16 * mi, 32, lane);
    v8f c = {};
    c = wmma_f16(a0, frag_b_nk_gf32(pout_w, 64, 0,  16 * ni, lane), c);
    c = wmma_f16(a1, frag_b_nk_gf32(pout_w, 64, 32, 16 * ni, lane), c);
#pragma unroll
    for (int r = 0; r < 8; ++r) {
      int m = 16 * mi + r + 8 * hg, n = 16 * ni + (lane & 15);
      outt[m * 65 + n] = c[r];
    }
  }
  __syncthreads();

  // coalesced writeout with residual re-read from workspace (L2-hot)
  for (int it = 0; it < 16; ++it) {
    int f = it * 256 + tid;
    int c = f >> 6, pix = f & 63;
    size_t ga = (((size_t)b * 64 + c) * 256 + r0 + (pix >> 3)) * 256 + c0 + (pix & 7);
    out[ga] = outt[pix * 65 + c] + x2[ga];
  }
}

extern "C" void kernel_launch(void* const* d_in, const int* in_sizes, int n_in,
                              void* d_out, int out_size, void* d_ws, size_t ws_size,
                              hipStream_t stream) {
  (void)in_sizes; (void)n_in; (void)out_size; (void)ws_size;
  const float* x      = (const float*)d_in[0];
  const float* ln_w   = (const float*)d_in[1];
  const float* ln_b   = (const float*)d_in[2];
  const float* qkv_w  = (const float*)d_in[3];   // [192][64]
  const float* out_w  = (const float*)d_in[4];   // [64][64]
  const float* cn_w   = (const float*)d_in[5];
  const float* cn_b   = (const float*)d_in[6];
  const float* pin_w  = (const float*)d_in[7];   // [128][64]
  const float* dw_w   = (const float*)d_in[8];   // [128][9]
  const float* pout_w = (const float*)d_in[9];   // [64][64]

  float* out      = (float*)d_out;               // [8,64,256,256] then attn [8192,1,64,64]
  float* attn_out = out + (size_t)ATTN_OFF;
  float* x2       = (float*)d_ws;                // 134,217,728 B workspace

  swin_attn_kernel<<<8192, 128, 0, stream>>>(x, ln_w, ln_b, qkv_w, out_w, x2, attn_out);
  swin_ffn_kernel<<<8192, 256, 0, stream>>>(x2, cn_w, cn_b, pin_w, dw_w, pout_w, out);
}
